// CosineAttention_15530601742609
// MI455X (gfx1250) — compile-verified, matrix operand order checked by reading, not run on hardware
//
#include <hip/hip_runtime.h>
#include <hip/hip_bf16.h>

// ---------------------------------------------------------------------------
// CosineAttention for MI455X (gfx1250, wave32, WMMA + Tensor Data Mover)
//   B=4, S=2048, D=1024 (full-width cosine attention, no head split)
//   bf16 operands + f32 WMMA accumulation via v_wmma_f32_16x16x32_bf16
//   bf16 GEMM tiles staged global->LDS by TDM (tensor_load_to_lds, TENSORcnt)
// ---------------------------------------------------------------------------

typedef __bf16 bf16;
typedef __attribute__((ext_vector_type(16))) __bf16 v16bf;
typedef __attribute__((ext_vector_type(8)))  __bf16 v8bf;
typedef __attribute__((ext_vector_type(4)))  __bf16 v4bf;
typedef __attribute__((ext_vector_type(8)))  float  v8f;
typedef __attribute__((ext_vector_type(4)))  unsigned int v4u;
typedef __attribute__((ext_vector_type(8)))  int v8i;
typedef __attribute__((ext_vector_type(4)))  int v4i;

#define LDS_STRIDE 40   // 32 data bf16 + 8 pad  -> 80 B rows, 16-B aligned chunks

#if __has_builtin(__builtin_amdgcn_tensor_load_to_lds) && \
    __has_builtin(__builtin_amdgcn_s_wait_tensorcnt)
#define USE_TDM 1
#else
#define USE_TDM 0
#endif

#if USE_TDM
// Issue one TDM 2D tile load: 128 rows x 32 bf16 (64 B/row), row stride K elems,
// into LDS with 16 B pad inserted after every 64 B -> LDS row stride 80 B,
// matching LDS_STRIDE. Per-wave instruction (EXEC ignored); call from one wave
// under a wave-uniform branch. Tracked by TENSORcnt.
__device__ inline void tdm_load_tile(const bf16* gsrc, const bf16* ldsDst, int K)
{
  const unsigned long long ga = (unsigned long long)(uintptr_t)gsrc;
  const unsigned lds = (unsigned)(uintptr_t)ldsDst;   // low 32 bits = LDS addr
  // D# group 0: count=1 | lds_addr | global_addr[56:0] | type=2
  v4u g0 = { 1u,
             lds,
             (unsigned)(ga & 0xFFFFFFFFu),
             (unsigned)((ga >> 32) & 0x01FFFFFFu) | 0x80000000u };
  // D# group 1: data_size=1(2B) | pad_enable | pad_interval=3(16 DW=64B)
  //             | pad_amount=3(4 DW=16B) ; tensor_dim0=K ; tensor_dim1=128 ;
  //             tile_dim0=32 ; tile_dim1=128 ; tensor_dim0_stride=K
  v8i g1 = { (int)((1u << 16) | (1u << 20) | (3u << 22) | (3u << 25)),
             (int)(((unsigned)K & 0xFFFFu) << 16),
             (int)((((unsigned)K >> 16) & 0xFFFFu) | (128u << 16)),
             (int)(32u << 16),
             (int)128,
             (int)K,
             0, 0 };
  v4i z4 = { 0, 0, 0, 0 };
#if __clang_major__ >= 23
  v8i z8 = { 0, 0, 0, 0, 0, 0, 0, 0 };
  __builtin_amdgcn_tensor_load_to_lds(g0, g1, z4, z4, z8, 0);
#else
  __builtin_amdgcn_tensor_load_to_lds(g0, g1, z4, z4, 0);
#endif
}
#endif  // USE_TDM

// Load one 16x32 bf16 A/B fragment from an LDS tile (row-major, stride LDS_STRIDE).
// ISA layout (7.12.2, 16-bit A 16x32): lanes0-15: m=lane, K {0..7,16..23};
// lanes16-31: m=lane-16, K {8..15,24..31}. Two 16-byte ds loads per fragment.
__device__ inline v16bf load_frag(const bf16* tile, int rowBase, int lane) {
  const int row = rowBase + (lane & 15);
  const int kb  = (lane >> 4) * 8;
  union { v16bf v; v8bf h[2]; } f;
  f.h[0] = *(const v8bf*)(tile + row * LDS_STRIDE + kb);
  f.h[1] = *(const v8bf*)(tile + row * LDS_STRIDE + kb + 16);
  return f.v;
}

// ---------------------------------------------------------------------------
// GEMM: C[M,N] = X[M,K] @ W[N,K]^T + bias   (f32 inputs, bf16 WMMA, f32 out)
// Block tile 128x128, 8 waves (4 along M x 2 along N), wave tile 32x64.
// Manual staging (needs f32->bf16 convert; TDM cannot convert).
// ---------------------------------------------------------------------------
__global__ __launch_bounds__(256)
void gemm_f32_xwT_bias(const float* __restrict__ X, const float* __restrict__ W,
                       const float* __restrict__ bias, float* __restrict__ C,
                       int M, int N, int K)
{
  __shared__ __align__(16) bf16 As[2][128 * LDS_STRIDE];
  __shared__ __align__(16) bf16 Bs[2][128 * LDS_STRIDE];

  const int tid  = threadIdx.x;
  const int lane = tid & 31, wave = tid >> 5;
  const int wm = wave & 3, wn = wave >> 2;
  const int mBase = blockIdx.y * 128;
  const int nBase = blockIdx.x * 128;

  const float* Xb = X + (long long)mBase * K;
  const float* Wb = W + (long long)nBase * K;

  v8f acc[2][4];
#pragma unroll
  for (int i = 0; i < 2; ++i)
#pragma unroll
    for (int j = 0; j < 4; ++j) acc[i][j] = {};

  auto stage = [&](int buf, int k0) {
#pragma unroll
    for (int s = 0; s < 4; ++s) {
      const int c = tid + s * 256;          // 1024 float4 chunks: 128 rows x 8
      const int row = c >> 3, q = c & 7;
      const float4 fa = *(const float4*)(Xb + (long long)row * K + k0 + q * 4);
      const float4 fb = *(const float4*)(Wb + (long long)row * K + k0 + q * 4);
      v4bf a4 = { (bf16)fa.x, (bf16)fa.y, (bf16)fa.z, (bf16)fa.w };
      v4bf b4 = { (bf16)fb.x, (bf16)fb.y, (bf16)fb.z, (bf16)fb.w };
      *(v4bf*)(&As[buf][row * LDS_STRIDE + q * 4]) = a4;
      *(v4bf*)(&Bs[buf][row * LDS_STRIDE + q * 4]) = b4;
    }
  };

  const int nK = K >> 5;
  stage(0, 0);
  __syncthreads();
  for (int kt = 0; kt < nK; ++kt) {
    const int cur = kt & 1;
    if (kt + 1 < nK) stage(cur ^ 1, (kt + 1) << 5);

    v16bf af[2], bfv[4];
#pragma unroll
    for (int mi = 0; mi < 2; ++mi) af[mi]  = load_frag(As[cur], wm * 32 + mi * 16, lane);
#pragma unroll
    for (int ni = 0; ni < 4; ++ni) bfv[ni] = load_frag(Bs[cur], wn * 64 + ni * 16, lane);

#pragma unroll
    for (int mi = 0; mi < 2; ++mi)
#pragma unroll
      for (int ni = 0; ni < 4; ++ni)
        acc[mi][ni] = __builtin_amdgcn_wmma_f32_16x16x32_bf16(
            false, af[mi], false, bfv[ni], (short)0, acc[mi][ni], false, false);
    __syncthreads();
  }

  // C/D layout: VGPR e -> row (lane<16 ? e : 8+e), col = lane&15
  const int mLane = (lane >= 16) ? 8 : 0;
  const int nCol  = lane & 15;
#pragma unroll
  for (int mi = 0; mi < 2; ++mi)
#pragma unroll
    for (int ni = 0; ni < 4; ++ni) {
      const int col = nBase + wn * 64 + ni * 16 + nCol;
      const float bv = bias ? bias[col] : 0.0f;
#pragma unroll
      for (int e = 0; e < 8; ++e) {
        const int row = mBase + wm * 32 + mi * 16 + mLane + e;
        C[(long long)row * N + col] = acc[mi][ni][e] + bv;
      }
    }
}

// ---------------------------------------------------------------------------
// GEMM: C[M,N] = scale * (A[M,K] @ Bm[N,K]^T)   (bf16 inputs, f32 out)
// Batched via blockIdx.z + strides. causal==1 skips tiles fully above diagonal.
// Tiles staged by the Tensor Data Mover (wave 0 issues, TENSORcnt-tracked),
// overlapped with WMMA on the other LDS buffer.
// ---------------------------------------------------------------------------
__global__ __launch_bounds__(256)
void gemm_bf16_abT(const bf16* __restrict__ A, const bf16* __restrict__ Bm,
                   float* __restrict__ C,
                   int M, int N, int K,
                   long long strideA, long long strideB, long long strideC,
                   float scale, int causal)
{
  __shared__ __align__(16) bf16 As[2][128 * LDS_STRIDE];
  __shared__ __align__(16) bf16 Bs[2][128 * LDS_STRIDE];

  const int tid  = threadIdx.x;
  const int lane = tid & 31, wave = tid >> 5;
  const int wm = wave & 3, wn = wave >> 2;
  const int mBase = blockIdx.y * 128;
  const int nBase = blockIdx.x * 128;
  const int b = blockIdx.z;
  if (causal && nBase > mBase + 127) return;   // tile entirely masked out

  const bf16* Ab = A  + (long long)b * strideA + (long long)mBase * K;
  const bf16* Bb = Bm + (long long)b * strideB + (long long)nBase * K;

  v8f acc[2][4];
#pragma unroll
  for (int i = 0; i < 2; ++i)
#pragma unroll
    for (int j = 0; j < 4; ++j) acc[i][j] = {};

  auto stage = [&](int buf, int k0) {
#if USE_TDM
    if (wave == 0) {               // wave-uniform: other waves scalar-skip
      tdm_load_tile(Ab + k0, &As[buf][0], K);
      tdm_load_tile(Bb + k0, &Bs[buf][0], K);
    }
#else
#pragma unroll
    for (int s = 0; s < 2; ++s) {
      const int c = tid + s * 256;          // 512 16-byte chunks: 128 rows x 4
      const int row = c >> 2, q = c & 3;
      v8bf va = *(const v8bf*)(Ab + (long long)row * K + k0 + q * 8);
      v8bf vb = *(const v8bf*)(Bb + (long long)row * K + k0 + q * 8);
      *(v8bf*)(&As[buf][row * LDS_STRIDE + q * 8]) = va;
      *(v8bf*)(&Bs[buf][row * LDS_STRIDE + q * 8]) = vb;
    }
#endif
  };
  auto stage_wait = [&]() {
#if USE_TDM
    if (wave == 0) __builtin_amdgcn_s_wait_tensorcnt(0);
#endif
  };

  const int nK = K >> 5;
  stage(0, 0);
  stage_wait();
  __syncthreads();
  for (int kt = 0; kt < nK; ++kt) {
    const int cur = kt & 1;
    if (kt + 1 < nK) stage(cur ^ 1, (kt + 1) << 5);   // DMA next, overlap compute

    v16bf af[2], bfv[4];
#pragma unroll
    for (int mi = 0; mi < 2; ++mi) af[mi]  = load_frag(As[cur], wm * 32 + mi * 16, lane);
#pragma unroll
    for (int ni = 0; ni < 4; ++ni) bfv[ni] = load_frag(Bs[cur], wn * 64 + ni * 16, lane);

#pragma unroll
    for (int mi = 0; mi < 2; ++mi)
#pragma unroll
      for (int ni = 0; ni < 4; ++ni)
        acc[mi][ni] = __builtin_amdgcn_wmma_f32_16x16x32_bf16(
            false, af[mi], false, bfv[ni], (short)0, acc[mi][ni], false, false);

    stage_wait();                  // next buffer landed (TENSORcnt -> 0)
    __syncthreads();               // publish to all waves
  }

  float* Cb = C + (long long)b * strideC;
  const int mLane = (lane >= 16) ? 8 : 0;
  const int nCol  = lane & 15;
#pragma unroll
  for (int mi = 0; mi < 2; ++mi)
#pragma unroll
    for (int ni = 0; ni < 4; ++ni) {
      const int col = nBase + wn * 64 + ni * 16 + nCol;
#pragma unroll
      for (int e = 0; e < 8; ++e) {
        const int row = mBase + wm * 32 + mi * 16 + mLane + e;
        Cb[(long long)row * N + col] = acc[mi][ni][e] * scale;
      }
    }
}

// ---------------------------------------------------------------------------
// Row L2-normalize (D=1024): Y = X / (||X||_2 + 1e-8), f32 -> bf16
// ---------------------------------------------------------------------------
__global__ __launch_bounds__(256)
void rownorm_to_bf16(const float* __restrict__ X, bf16* __restrict__ Y, int D)
{
  __shared__ float red[256];
  const long long row = blockIdx.x;
  const int t = threadIdx.x;
  const float4 v = *(const float4*)(X + row * D + t * 4);
  red[t] = v.x * v.x + v.y * v.y + v.z * v.z + v.w * v.w;
  __syncthreads();
  for (int s = 128; s > 0; s >>= 1) {
    if (t < s) red[t] += red[t + s];
    __syncthreads();
  }
  const float inv = 1.0f / (sqrtf(red[0]) + 1e-8f);
  v4bf o = { (bf16)(v.x * inv), (bf16)(v.y * inv),
             (bf16)(v.z * inv), (bf16)(v.w * inv) };
  *(v4bf*)(Y + row * D + t * 4) = o;
}

// ---------------------------------------------------------------------------
// V[b,s,d] f32 -> Vt[b,d,s] bf16 (so PV GEMM becomes row-major A*B^T)
// ---------------------------------------------------------------------------
__global__ void transpose_to_bf16(const float* __restrict__ V, bf16* __restrict__ Vt,
                                  int Sdim, int Ddim)
{
  __shared__ bf16 tile[32][33];
  const int b = blockIdx.z;
  const int tBase = blockIdx.x * 32;
  const int dBase = blockIdx.y * 32;
  const float* Vb = V  + (long long)b * Sdim * Ddim;
  bf16*       Vtb = Vt + (long long)b * Sdim * Ddim;
  const int tx = threadIdx.x, ty = threadIdx.y;   // (32, 8)
#pragma unroll
  for (int i = 0; i < 4; ++i) {
    const int tr = ty + i * 8;
    tile[tr][tx] = (bf16)Vb[(long long)(tBase + tr) * Ddim + dBase + tx];
  }
  __syncthreads();
#pragma unroll
  for (int i = 0; i < 4; ++i) {
    const int dr = ty + i * 8;
    Vtb[(long long)(dBase + dr) * Sdim + tBase + tx] = tile[tx][dr];
  }
}

// ---------------------------------------------------------------------------
// Causal row softmax: P[b,i,:] = softmax(scores[b,i,0..i]) (bf16), 0 above diag.
// ---------------------------------------------------------------------------
__global__ __launch_bounds__(256)
void causal_softmax_rows(const float* __restrict__ Sc, bf16* __restrict__ P, int Sdim)
{
  __shared__ float red[256];
  const int i = blockIdx.x;
  const int b = blockIdx.y;
  const float* srow = Sc + ((long long)b * Sdim + i) * Sdim;
  bf16*        prow = P  + ((long long)b * Sdim + i) * Sdim;
  const int t = threadIdx.x;

  float s[8];
#pragma unroll
  for (int k = 0; k < 8; ++k) {
    const int j = t + k * 256;
    s[k] = (j <= i) ? srow[j] : -__builtin_inff();
  }
  float m = s[0];
#pragma unroll
  for (int k = 1; k < 8; ++k) m = fmaxf(m, s[k]);
  red[t] = m;
  __syncthreads();
  for (int w = 128; w > 0; w >>= 1) {
    if (t < w) red[t] = fmaxf(red[t], red[t + w]);
    __syncthreads();
  }
  m = red[0];
  __syncthreads();

  float lsum = 0.0f;
#pragma unroll
  for (int k = 0; k < 8; ++k) { s[k] = __expf(s[k] - m); lsum += s[k]; }
  red[t] = lsum;
  __syncthreads();
  for (int w = 128; w > 0; w >>= 1) {
    if (t < w) red[t] += red[t + w];
    __syncthreads();
  }
  const float inv = 1.0f / red[0];
#pragma unroll
  for (int k = 0; k < 8; ++k) {
    const int j = t + k * 256;
    prow[j] = (bf16)(s[k] * inv);
  }
}

// ---------------------------------------------------------------------------
// Host-side orchestration
// ---------------------------------------------------------------------------
extern "C" void kernel_launch(void* const* d_in, const int* in_sizes, int n_in,
                              void* d_out, int out_size, void* d_ws, size_t ws_size,
                              hipStream_t stream)
{
  (void)in_sizes; (void)n_in; (void)out_size; (void)ws_size;

  constexpr int B = 4, S = 2048, D = 1024;
  constexpr float SCALE = 0.03125f;  // 1024^-0.5
  const size_t BSD = (size_t)B * S * D;   // 8,388,608 elems
  const size_t BSS = (size_t)B * S * S;   // 16,777,216 elems

  const float* q  = (const float*)d_in[0];
  const float* k  = (const float*)d_in[1];
  const float* v  = (const float*)d_in[2];
  const float* Wq = (const float*)d_in[3];
  const float* bq = (const float*)d_in[4];
  const float* Wk = (const float*)d_in[5];
  const float* bk = (const float*)d_in[6];
  const float* Wv = (const float*)d_in[7];
  const float* bv = (const float*)d_in[8];
  const float* Wo = (const float*)d_in[9];
  const float* bo = (const float*)d_in[10];
  // d_in[11] attn_mask (causal, reproduced analytically), d_in[12] padding (all false)

  // Workspace carve-up
  char* ws = (char*)d_ws;
  size_t off = 0;
  auto carve = [&](size_t bytes) {
    void* p = ws + off;
    off += (bytes + 255) & ~(size_t)255;
    return p;
  };
  float* Qf = (float*)carve(BSD * sizeof(float));
  float* Kf = (float*)carve(BSD * sizeof(float));
  float* Vf = (float*)carve(BSD * sizeof(float));
  bf16*  Qn = (bf16*)carve(BSD * sizeof(bf16));
  bf16*  Kn = (bf16*)carve(BSD * sizeof(bf16));
  bf16*  Vt = (bf16*)carve(BSD * sizeof(bf16));
  bf16*  P  = (bf16*)carve(BSS * sizeof(bf16));
  float* Sc = (float*)carve(BSS * sizeof(float));
  float* Ao = Sc;  // attention-out reuses scores buffer (disjoint in time)

  const dim3 blk(256);
  const dim3 gProj(D / 128, (B * S) / 128);          // (8, 64)

  // 1) Projections: Q/K/V = x @ W^T + b
  gemm_f32_xwT_bias<<<gProj, blk, 0, stream>>>(q, Wq, bq, Qf, B * S, D, D);
  gemm_f32_xwT_bias<<<gProj, blk, 0, stream>>>(k, Wk, bk, Kf, B * S, D, D);
  gemm_f32_xwT_bias<<<gProj, blk, 0, stream>>>(v, Wv, bv, Vf, B * S, D, D);

  // 2) L2 normalize Q, K over full d_model; 3) V -> V^T bf16
  rownorm_to_bf16<<<dim3(B * S), blk, 0, stream>>>(Qf, Qn, D);
  rownorm_to_bf16<<<dim3(B * S), blk, 0, stream>>>(Kf, Kn, D);
  transpose_to_bf16<<<dim3(S / 32, D / 32, B), dim3(32, 8), 0, stream>>>(Vf, Vt, S, D);

  // 4) scores = SCALE * Qn @ Kn^T (causal tiles above diagonal skipped)
  gemm_bf16_abT<<<dim3(S / 128, S / 128, B), blk, 0, stream>>>(
      Qn, Kn, Sc, S, S, D,
      (long long)S * D, (long long)S * D, (long long)S * S, SCALE, 1);

  // 5) P = causal softmax(scores), bf16
  causal_softmax_rows<<<dim3(S, B), blk, 0, stream>>>(Sc, P, S);

  // 6) attn_out = P @ V  (via V^T rows)
  gemm_bf16_abT<<<dim3(D / 128, S / 128, B), blk, 0, stream>>>(
      P, Vt, Ao, S, D, S,
      (long long)S * S, (long long)S * D, (long long)S * D, 1.0f, 0);

  // 7) out = attn_out @ Wo^T + bo
  gemm_f32_xwT_bias<<<gProj, blk, 0, stream>>>(Ao, Wo, bo, (float*)d_out, B * S, D, D);
}
